// GATLayer_69647189672500
// MI455X (gfx1250) — matrix operand order repaired and checked by reading, not compiled, not run on hardware
//
#include <hip/hip_runtime.h>
#include <hip/hip_bf16.h>

#define GN 4096
#define GIN_F 256
#define GOUT_F 64
#define GH 4
#define NEG_SLOPE 0.2f

typedef __attribute__((ext_vector_type(2)))  float    v2f;
typedef __attribute__((ext_vector_type(4)))  float    f4;
typedef __attribute__((ext_vector_type(8)))  float    v8f;
typedef __attribute__((ext_vector_type(16))) _Float16 v16h;
typedef __attribute__((ext_vector_type(4)))  unsigned u32x4;
typedef __attribute__((ext_vector_type(8)))  int      i32x8;
typedef __attribute__((ext_vector_type(4)))  int      i32x4;

// ---------------------------------------------------------------------------
// Kernel 1: Wh = x @ W per head via V_WMMA_F32_16X16X4_F32 (fp32 precision).
// Also computes s_src[n][h], s_dstT[h][n] and writes Whh in fp16, transposed
// to [h][o][j] so attention B tiles are two 16B runs along j per lane.
// ---------------------------------------------------------------------------
__global__ __launch_bounds__(32) void gat_wh_kernel(
    const float* __restrict__ x, const float* __restrict__ W,
    const float* __restrict__ a, _Float16* __restrict__ Whh,
    float* __restrict__ s_src, float* __restrict__ s_dstT)
{
    const int lane = threadIdx.x;
    const int nn  = lane & 15;
    const int hlf = lane >> 4;
    const int i0  = blockIdx.x * 16;

    v8f c[GH][4];
    #pragma unroll
    for (int h = 0; h < GH; ++h)
        #pragma unroll
        for (int o = 0; o < 4; ++o) {
            v8f z = {0.f,0.f,0.f,0.f,0.f,0.f,0.f,0.f};
            c[h][o] = z;
        }

    const float* xrow = x + (size_t)(i0 + nn) * GIN_F;

    for (int kc = 0; kc < GIN_F; kc += 4) {
        v2f av;
        av.x = xrow[kc + 2*hlf + 0];
        av.y = xrow[kc + 2*hlf + 1];
        #pragma unroll
        for (int h = 0; h < GH; ++h) {
            const float* wb = W + ((size_t)h * GIN_F + kc + 2*hlf) * GOUT_F;
            #pragma unroll
            for (int o = 0; o < 4; ++o) {
                v2f bv;
                bv.x = wb[o*16 + nn];
                bv.y = wb[GOUT_F + o*16 + nn];
                c[h][o] = __builtin_amdgcn_wmma_f32_16x16x4_f32(
                    false, av, false, bv, (short)0, c[h][o], false, false);
            }
        }
    }

    const int mrow0 = i0 + 8*hlf;
    #pragma unroll
    for (int h = 0; h < GH; ++h) {
        float psrc[8], pdst[8];
        #pragma unroll
        for (int r = 0; r < 8; ++r) { psrc[r] = 0.f; pdst[r] = 0.f; }

        #pragma unroll
        for (int o = 0; o < 4; ++o) {
            const float asv = a[(size_t)h*2*GOUT_F + o*16 + nn];
            const float adv = a[(size_t)h*2*GOUT_F + GOUT_F + o*16 + nn];
            union { f4 v; _Float16 hh[8]; } pk;
            #pragma unroll
            for (int r = 0; r < 8; ++r) {
                float cv = c[h][o][r];
                pk.hh[r] = (_Float16)cv;
                psrc[r] += cv * asv;
                pdst[r] += cv * adv;
            }
            _Float16* dst = Whh + (size_t)(h*GOUT_F + o*16 + nn) * GN + mrow0;
            *(f4*)dst = pk.v;
        }
        #pragma unroll
        for (int r = 0; r < 8; ++r) {
            float vs = psrc[r], vd = pdst[r];
            vs += __shfl_xor(vs, 1, 32);  vd += __shfl_xor(vd, 1, 32);
            vs += __shfl_xor(vs, 2, 32);  vd += __shfl_xor(vd, 2, 32);
            vs += __shfl_xor(vs, 4, 32);  vd += __shfl_xor(vd, 4, 32);
            vs += __shfl_xor(vs, 8, 32);  vd += __shfl_xor(vd, 8, 32);
            if (nn == 0) {
                s_src[(size_t)(mrow0 + r) * GH + h] = vs;
                s_dstT[(size_t)h * GN + mrow0 + r]  = vd;
            }
        }
    }
}

// ---------------------------------------------------------------------------
// Kernel 2: Smax[h] = max_j s_dst[j][h]
// ---------------------------------------------------------------------------
__global__ __launch_bounds__(128) void gat_smax_kernel(
    const float* __restrict__ s_dstT, float* __restrict__ Smax)
{
    const int h = threadIdx.x >> 5, lane = threadIdx.x & 31;
    float mx = -1e30f;
    for (int j = lane; j < GN; j += 32) mx = fmaxf(mx, s_dstT[(size_t)h*GN + j]);
    #pragma unroll
    for (int m = 16; m >= 1; m >>= 1) mx = fmaxf(mx, __shfl_xor(mx, m, 32));
    if (lane == 0) Smax[h] = mx;
}

// ---------------------------------------------------------------------------
// TDM helpers: build D# groups per CDNA5 ISA.
// 2D tile: tile_dim0 (contiguous) x tile_dim1 rows, row stride stride0,
// all in data_size-element units. Groups 2/3 zero (<=2D tensor).
// Toolchain uses the 6-arg builtin: (g0, g1, g2, g3, g4, cpol).
// ---------------------------------------------------------------------------
__device__ __forceinline__ void tdm_load_2d(
    unsigned lds_off, unsigned long long gaddr, int ds_code,
    unsigned tdim0, unsigned tdim1, unsigned tile0, unsigned tile1,
    unsigned stride0)
{
    u32x4 g0;
    g0[0] = 1u;                                        // count=1, user mode
    g0[1] = lds_off;                                   // lds_addr (bytes)
    g0[2] = (unsigned)(gaddr & 0xFFFFFFFFull);         // global_addr[31:0]
    g0[3] = (unsigned)((gaddr >> 32) & 0x1FFFFFFull)   // global_addr[56:32]
          | (2u << 30);                                // type=2 (image)
    i32x8 g1;
    g1[0] = ds_code << 16;                             // data_size, no multicast
    g1[1] = (int)(tdim0 << 16);                        // tensor_dim0[15:0]
    g1[2] = (int)((tdim0 >> 16) | (tdim1 << 16));      // tdim0 hi | tdim1 lo
    g1[3] = (int)((tdim1 >> 16) | (tile0 << 16));      // tdim1 hi | tile_dim0
    g1[4] = (int)(tile1 & 0xFFFF);                     // tile_dim1 (tile_dim2=0)
    g1[5] = (int)stride0;                              // tensor_dim0_stride lo
    g1[6] = 0;                                         // stride hi | dim1_stride
    g1[7] = 0;
    i32x4 gz  = {0, 0, 0, 0};
    i32x8 gz8 = {0, 0, 0, 0, 0, 0, 0, 0};
    __builtin_amdgcn_tensor_load_to_lds(g0, g1, gz, gz, gz8, 0);
}

__device__ __forceinline__ unsigned lds_offset_of(const void* p)
{
    // LDS aperture flat addresses carry the LDS byte offset in the low 32 bits
    return (unsigned)(unsigned long long)p;
}

// ---------------------------------------------------------------------------
// Kernel 3: fused masked-softmax attention * Wh.
// Workgroup = 4 waves = 64 i-rows, walking the same 32-wide j chunk in
// lockstep. adj tile (64x32 f32) and Wh tile (256x32 f16) are staged into
// LDS by the Tensor Data Mover (double buffered, TENSORcnt + barriers).
// p' is built directly in the WMMA f16 A-matrix lane layout.
// ---------------------------------------------------------------------------
__global__ __launch_bounds__(128) void gat_attn_kernel(
    const float* __restrict__ adj, const float* __restrict__ s_src,
    const float* __restrict__ s_dstT, const float* __restrict__ Smax,
    const _Float16* __restrict__ Whh, float* __restrict__ part,
    float* __restrict__ lpart, int seglen)
{
    __shared__ float    sAdj[2][64 * 32];     // 2 x 8 KB
    __shared__ _Float16 sWhh[2][256 * 32];    // 2 x 16 KB

    const int tid  = threadIdx.x;
    const int w    = tid >> 5;                // wave in workgroup (0..3)
    const int lane = tid & 31;
    const int nn   = lane & 15;
    const int hlf  = lane >> 4;
    const int ib   = blockIdx.x * 64;         // workgroup row base
    const int i0   = ib + w * 16;             // this wave's i-tile
    const int seg  = blockIdx.y;
    const int row  = i0 + nn;

    // scalar-uniform wave id so the TDM guard becomes a scalar branch
    const int wv = __builtin_amdgcn_readfirstlane(w);

    float si[GH], mh[GH];
    #pragma unroll
    for (int h = 0; h < GH; ++h) {
        si[h] = s_src[(size_t)row * GH + h];
        float e = si[h] + Smax[h];
        mh[h] = e > 0.f ? e : NEG_SLOPE * e;  // row-wise bound of masked max
    }

    v8f c[GH][4];
    #pragma unroll
    for (int h = 0; h < GH; ++h)
        #pragma unroll
        for (int o = 0; o < 4; ++o) {
            v8f z = {0.f,0.f,0.f,0.f,0.f,0.f,0.f,0.f};
            c[h][o] = z;
        }
    float lsum[GH] = {0.f, 0.f, 0.f, 0.f};

    const int jbeg  = seg * seglen;
    const int iters = seglen / 32;

    // prologue: DMA first chunk into buffer 0
    if (wv == 0) {
        tdm_load_2d(lds_offset_of(&sAdj[0][0]),
                    (unsigned long long)(adj + (size_t)ib * GN + jbeg),
                    /*f32*/2, GN, GN, 32, 64, GN);
        tdm_load_2d(lds_offset_of(&sWhh[0][0]),
                    (unsigned long long)(Whh + jbeg),
                    /*f16*/1, GN, GH*GOUT_F, 32, GH*GOUT_F, GN);
    }

    for (int it = 0; it < iters; ++it) {
        const int jc = jbeg + it * 32;
        const int p  = it & 1;

        if (wv == 0) {
            if (it + 1 < iters) {     // issue next pair, wait for current pair
                tdm_load_2d(lds_offset_of(&sAdj[1 - p][0]),
                            (unsigned long long)(adj + (size_t)ib * GN + jc + 32),
                            2, GN, GN, 32, 64, GN);
                tdm_load_2d(lds_offset_of(&sWhh[1 - p][0]),
                            (unsigned long long)(Whh + jc + 32),
                            1, GN, GH*GOUT_F, 32, GH*GOUT_F, GN);
                __builtin_amdgcn_s_wait_tensorcnt(2);
            } else {
                __builtin_amdgcn_s_wait_tensorcnt(0);
            }
        }
        __syncthreads();              // buffer p ready for everyone

        const int kb0 = jc + 8*hlf;          // A slots 0..7  -> K = kb0+t
        const int kb1 = jc + 16 + 8*hlf;     // A slots 8..15 -> K = kb1+t
        const float* ar = &sAdj[p][(w*16 + nn) * 32];
        f4 q0 = *(const f4*)(ar + 8*hlf);
        f4 q1 = *(const f4*)(ar + 8*hlf + 4);
        f4 q2 = *(const f4*)(ar + 16 + 8*hlf);
        f4 q3 = *(const f4*)(ar + 16 + 8*hlf + 4);

        float ae[16];                 // effective adjacency with self-loop
        #pragma unroll
        for (int t = 0; t < 4; ++t) {
            ae[t]      = (q0[t] != 0.f || (kb0 + t)     == row) ? 1.f : 0.f;
            ae[4 + t]  = (q1[t] != 0.f || (kb0 + 4 + t) == row) ? 1.f : 0.f;
            ae[8 + t]  = (q2[t] != 0.f || (kb1 + t)     == row) ? 1.f : 0.f;
            ae[12 + t] = (q3[t] != 0.f || (kb1 + 4 + t) == row) ? 1.f : 0.f;
        }

        #pragma unroll
        for (int h = 0; h < GH; ++h) {
            const float* sd = s_dstT + (size_t)h * GN;
            f4 d0 = *(const f4*)(sd + kb0);
            f4 d1 = *(const f4*)(sd + kb0 + 4);
            f4 d2 = *(const f4*)(sd + kb1);
            f4 d3 = *(const f4*)(sd + kb1 + 4);

            union { v16h v; _Float16 e16[16]; } A;
            float ls = 0.f;
            #pragma unroll
            for (int t = 0; t < 4; ++t) {
                float e0 = si[h] + d0[t]; e0 = e0 > 0.f ? e0 : NEG_SLOPE*e0;
                float p0 = ae[t] * __expf(e0 - mh[h]); ls += p0; A.e16[t] = (_Float16)p0;
                float e1 = si[h] + d1[t]; e1 = e1 > 0.f ? e1 : NEG_SLOPE*e1;
                float p1 = ae[4+t] * __expf(e1 - mh[h]); ls += p1; A.e16[4+t] = (_Float16)p1;
                float e2 = si[h] + d2[t]; e2 = e2 > 0.f ? e2 : NEG_SLOPE*e2;
                float p2 = ae[8+t] * __expf(e2 - mh[h]); ls += p2; A.e16[8+t] = (_Float16)p2;
                float e3 = si[h] + d3[t]; e3 = e3 > 0.f ? e3 : NEG_SLOPE*e3;
                float p3 = ae[12+t] * __expf(e3 - mh[h]); ls += p3; A.e16[12+t] = (_Float16)p3;
            }
            lsum[h] += ls;

            #pragma unroll
            for (int o = 0; o < 4; ++o) {
                // B tile from LDS: slots 0..7 K=8*hlf+s, slots 8..15 K=16+8*hlf+(s-8)
                const _Float16* bp = &sWhh[p][(h*GOUT_F + o*16 + nn) * 32 + 8*hlf];
                union { f4 q[2]; v16h v; } Bu;
                Bu.q[0] = *(const f4*)bp;
                Bu.q[1] = *(const f4*)(bp + 16);
                v16h B = Bu.v;
                c[h][o] = __builtin_amdgcn_wmma_f32_16x16x32_f16(
                    false, A.v, false, B, (short)0, c[h][o], false, false);
            }
        }
        __syncthreads();              // buffer p free for next DMA
    }

    float lf[GH];
    #pragma unroll
    for (int h = 0; h < GH; ++h) lf[h] = lsum[h] + __shfl_xor(lsum[h], 16, 32);

    float* pc = part + (size_t)seg * GN * (GH * GOUT_F);
    #pragma unroll
    for (int h = 0; h < GH; ++h)
        #pragma unroll
        for (int o = 0; o < 4; ++o)
            #pragma unroll
            for (int r = 0; r < 8; ++r) {
                const int rr = i0 + 8*hlf + r;
                pc[(size_t)rr * (GH*GOUT_F) + h*GOUT_F + o*16 + nn] = c[h][o][r];
            }
    if (hlf == 0) {
        float* lp = lpart + (size_t)seg * GN * GH;
        #pragma unroll
        for (int h = 0; h < GH; ++h) lp[(size_t)row * GH + h] = lf[h];
    }
}

// ---------------------------------------------------------------------------
// Kernel 4: sum j-segment partials and normalize by l.
// ---------------------------------------------------------------------------
__global__ __launch_bounds__(256) void gat_reduce_kernel(
    const float* __restrict__ part, const float* __restrict__ lpart,
    float* __restrict__ out, int segs)
{
    const int i = blockIdx.x, f = threadIdx.x, h = f >> 6;
    float sp = 0.f, sl = 0.f;
    for (int s = 0; s < segs; ++s) {
        sp += part[(size_t)s * GN * (GH*GOUT_F) + (size_t)i * (GH*GOUT_F) + f];
        sl += lpart[(size_t)s * GN * GH + (size_t)i * GH + h];
    }
    out[(size_t)i * (GH*GOUT_F) + f] = sp / sl;
}

// ---------------------------------------------------------------------------
extern "C" void kernel_launch(void* const* d_in, const int* in_sizes, int n_in,
                              void* d_out, int out_size, void* d_ws, size_t ws_size,
                              hipStream_t stream)
{
    const float* x   = (const float*)d_in[0];   // (4096,256)
    const float* adj = (const float*)d_in[1];   // (4096,4096)
    const float* W   = (const float*)d_in[2];   // (4,256,64)
    const float* a   = (const float*)d_in[3];   // (4,128,1)
    float* out = (float*)d_out;                 // (4096,256) fp32

    char* ws = (char*)d_ws;
    size_t off = 0;
    auto alloc = [&](size_t bytes) -> char* {
        char* p = ws + off;
        off = (off + bytes + 255) & ~(size_t)255;
        return p;
    };
    _Float16* Whh   = (_Float16*)alloc((size_t)GH * GOUT_F * GN * sizeof(_Float16));
    float*    s_src = (float*)alloc((size_t)GN * GH * sizeof(float));
    float*    s_dstT= (float*)alloc((size_t)GN * GH * sizeof(float));
    float*    Smax  = (float*)alloc(GH * sizeof(float));

    const size_t per_seg = (size_t)GN * GH * GOUT_F * sizeof(float)
                         + (size_t)GN * GH * sizeof(float) + 512;
    int segs = 8;
    while (segs > 1 && off + (size_t)segs * per_seg > ws_size) segs >>= 1;
    float* part  = (float*)alloc((size_t)segs * GN * GH * GOUT_F * sizeof(float));
    float* lpart = (float*)alloc((size_t)segs * GN * GH * sizeof(float));
    const int seglen = GN / segs;

    gat_wh_kernel<<<GN / 16, 32, 0, stream>>>(x, W, a, Whh, s_src, s_dstT);
    gat_smax_kernel<<<1, 128, 0, stream>>>(s_dstT, Smax);
    gat_attn_kernel<<<dim3(GN / 64, segs), 128, 0, stream>>>(
        adj, s_src, s_dstT, Smax, Whh, part, lpart, seglen);
    gat_reduce_kernel<<<GN, 256, 0, stream>>>(part, lpart, out, segs);
}
